// PatchAttentionBlock_84559316123842
// MI455X (gfx1250) — matrix operand order repaired and checked by reading, not compiled, not run on hardware
//
#include <hip/hip_runtime.h>
#include <hip/hip_bf16.h>

// ---------- types ----------
typedef _Float16 h16;
typedef __attribute__((ext_vector_type(8)))  _Float16 h8;
typedef __attribute__((ext_vector_type(16))) _Float16 v16h;
typedef __attribute__((ext_vector_type(8)))  float    v8f;

// ---------- problem constants ----------
#define BB 16
#define CC 4
#define SS 1024
#define DD 512
#define HH 8
#define HD 64
#define FF 2048
#define MTOT (BB*SS)          // 16384 rows

// ---------- CDNA5 async global->LDS (inline asm; ASYNCcnt-tracked) ----------
// Copies 128 contiguous bytes from global to LDS via 8 async b128 transfers.
// ISA 08_async_tensor §4.4: INST_OFFSET is added to BOTH the LDS and global
// addresses, so one register pair + immediate offsets covers the whole row.
__device__ __forceinline__ void async_cp128B(h16* lds, const h16* g) {
    unsigned l = (unsigned)(unsigned long long)(void*)lds;  // LDS addr = low 32 bits
    asm volatile(
        "global_load_async_to_lds_b128 %0, %1, off\n\t"
        "global_load_async_to_lds_b128 %0, %1, off offset:16\n\t"
        "global_load_async_to_lds_b128 %0, %1, off offset:32\n\t"
        "global_load_async_to_lds_b128 %0, %1, off offset:48\n\t"
        "global_load_async_to_lds_b128 %0, %1, off offset:64\n\t"
        "global_load_async_to_lds_b128 %0, %1, off offset:80\n\t"
        "global_load_async_to_lds_b128 %0, %1, off offset:96\n\t"
        "global_load_async_to_lds_b128 %0, %1, off offset:112"
        :: "v"(l), "v"(g) : "memory");
}
// wait until at most N async ops outstanding (in-order completion per wave)
#define ASYNC_WAIT_PREV() asm volatile("s_wait_asynccnt 0x8" ::: "memory")
#define ASYNC_WAIT_ALL()  asm volatile("s_wait_asynccnt 0x0" ::: "memory")

// ---------- WMMA fragment helpers ----------
// A-matrix 16x32 f16 layout (ISA 7.12.2): lane<16 holds K = kb..kb+7 and kb+16..kb+23
// with kb = (lane>=16 ? 8 : 0).  K-contiguous source required.
__device__ __forceinline__ v16h load_afrag(const h16* p /* &row[k0] for this lane */) {
    h8 a = *(const h8*)p;
    h8 b = *(const h8*)(p + 16);
    return __builtin_shufflevector(a, b, 0,1,2,3,4,5,6,7,8,9,10,11,12,13,14,15);
}
// B-matrix 32x16 f16 layout: lane holds K = kb..kb+7 and kb+16..kb+23, kb = (lane>=16 ? 8 : 0).
__device__ __forceinline__ v16h load_bfrag(const h16* p /* &col[k0] for this lane */) {
    h8 a = *(const h8*)p;
    h8 b = *(const h8*)(p + 16);
    return __builtin_shufflevector(a, b, 0,1,2,3,4,5,6,7,8,9,10,11,12,13,14,15);
}
__device__ __forceinline__ v8f wmma_f16(v16h a, v16h b, v8f c) {
    return __builtin_amdgcn_wmma_f32_16x16x32_f16(false, a, false, b, (short)0, c, false, false);
}

// ---------- weight transpose + f16 convert:  Wt[n*K+k] = W[k*N+n] ----------
__global__ void wcvt_kernel(const float* __restrict__ W, h16* __restrict__ Wt, int K, int N) {
    int i = blockIdx.x * 256 + threadIdx.x;
    if (i >= K * N) return;
    int k = i / N, n = i - k * N;
    Wt[(size_t)n * K + k] = (h16)W[i];
}

// ---------- LayerNorm: one wave per row, f32 in (strided by batch), f16 out ----------
__global__ __launch_bounds__(256)
void ln_kernel(const float* __restrict__ x, long chanStride,
               const float* __restrict__ g, const float* __restrict__ bia,
               h16* __restrict__ out) {
    int lane = threadIdx.x & 31, wid = threadIdx.x >> 5;
    int row  = blockIdx.x * 8 + wid;
    long b = row >> 10, s = row & 1023;
    const float* p = x + b * chanStride + s * DD;
    float v[16];
    float sum = 0.f;
#pragma unroll
    for (int i = 0; i < 16; ++i) { v[i] = p[lane + i * 32]; sum += v[i]; }
#pragma unroll
    for (int o = 16; o; o >>= 1) sum += __shfl_xor(sum, o, 32);
    float mu = sum * (1.0f / DD);
    float vs = 0.f;
#pragma unroll
    for (int i = 0; i < 16; ++i) { float d = v[i] - mu; vs += d * d; }
#pragma unroll
    for (int o = 16; o; o >>= 1) vs += __shfl_xor(vs, o, 32);
    float inv = rsqrtf(vs * (1.0f / DD) + 1e-5f);
    h16* op = out + (size_t)row * DD;
#pragma unroll
    for (int i = 0; i < 16; ++i) {
        int c = lane + i * 32;
        op[c] = (h16)((v[i] - mu) * inv * g[c] + bia[c]);
    }
}

// ---------- generic WMMA GEMM:  C[M,N] = A[M,K](f16) * Bt[N,K]^T(f16) + bias ----------
// 256 threads = 8 waves; block tile 128x128, K-step 64, double-buffered async LDS
// staging; wave tile 32x64 (2x4 frags) -> 16 WMMAs between barriers.
#define TM 128
#define TN 128
#define TK 64
#define LDP 72   // LDS halves per row (64 + 8 pad; 144B row, 16B aligned)

#define EPI_F16      0   // out f16 = acc + bias
#define EPI_GELU_F16 1   // out f16 = gelu(acc + bias)
#define EPI_RES_F32  2   // out f32 = acc + bias + resid   (batch-strided in & out)

template<int EPI>
__global__ __launch_bounds__(256)
void gemm_wmma(const h16* __restrict__ A, const h16* __restrict__ Bt,
               const float* __restrict__ bias,
               const float* __restrict__ resid, long residChanStride,
               long outChanStride,
               h16* __restrict__ outH, float* __restrict__ outF,
               int M, int N, int K) {
    __shared__ h16 As[2][TM * LDP];
    __shared__ h16 Bs[2][TN * LDP];
    const int t = threadIdx.x;
    const int lane = t & 31, wid = t >> 5;
    const int wm = wid & 3, wn = wid >> 2;           // 4x2 wave grid
    const int m0 = blockIdx.y * TM;
    const int n0 = blockIdx.x * TN;
    const int akb = ((lane >> 4) << 3);              // 0 or 8  (A frag)
    const int bkb = ((lane >> 4) << 3);              // 0 or 8  (B frag)
    const int lrow = lane & 15;

    // stage one 128x64 A slab + 128x64 B slab into LDS buffer `buf` (async)
    auto stage = [&](int buf, int kt) {
        if (t < 128) {
            async_cp128B(&As[buf][t * LDP], A + (size_t)(m0 + t) * K + kt);
        } else {
            int r = t - 128;
            async_cp128B(&Bs[buf][r * LDP], Bt + (size_t)(n0 + r) * K + kt);
        }
    };

    v8f acc[2][4] = {};
    const int nk = K / TK;

    stage(0, 0);
    for (int ki = 0; ki < nk; ++ki) {
        const int cur = ki & 1;
        if (ki + 1 < nk) {
            stage(cur ^ 1, (ki + 1) * TK);
            ASYNC_WAIT_PREV();   // 8 new loads in flight; previous stage complete
        } else {
            ASYNC_WAIT_ALL();
        }
        __syncthreads();         // everyone's current-stage data visible in LDS

#pragma unroll
        for (int kk = 0; kk < 2; ++kk) {
            v16h af[2], bf[4];
#pragma unroll
            for (int fm = 0; fm < 2; ++fm)
                af[fm] = load_afrag(&As[cur][(wm * 32 + fm * 16 + lrow) * LDP + kk * 32 + akb]);
#pragma unroll
            for (int fn = 0; fn < 4; ++fn)
                bf[fn] = load_bfrag(&Bs[cur][(wn * 64 + fn * 16 + lrow) * LDP + kk * 32 + bkb]);
#pragma unroll
            for (int fm = 0; fm < 2; ++fm)
#pragma unroll
                for (int fn = 0; fn < 4; ++fn)
                    acc[fm][fn] = wmma_f16(af[fm], bf[fn], acc[fm][fn]);
        }
        __syncthreads();         // all reads of buf[cur] done before it is restaged
    }

    // epilogue: C layout — lane = n + (m>=8)*16 ; vgpr r = m&7
    const int hi8 = ((lane >> 4) << 3);
#pragma unroll
    for (int fn = 0; fn < 4; ++fn) {
        int col = n0 + wn * 64 + fn * 16 + lrow;
        float bv = bias[col];
#pragma unroll
        for (int fm = 0; fm < 2; ++fm) {
#pragma unroll
            for (int r = 0; r < 8; ++r) {
                int row = m0 + wm * 32 + fm * 16 + r + hi8;
                float v = acc[fm][fn][r] + bv;
                if (EPI == EPI_F16) {
                    outH[(size_t)row * N + col] = (h16)v;
                } else if (EPI == EPI_GELU_F16) {
                    v = 0.5f * v * (1.0f + erff(v * 0.70710678118f));
                    outH[(size_t)row * N + col] = (h16)v;
                } else { // EPI_RES_F32
                    long b = row >> 10, s = row & 1023;
                    float rv = resid[b * residChanStride + s * (long)N + col];
                    outF[b * outChanStride + s * (long)N + col] = v + rv;
                }
            }
        }
    }
}

// ---------- RoPE(2D) + QKV split + V transpose ----------
// qkv f16 [M,1536] -> qr,kr f16 [B,H,S,HD], vt f16 [B,H,HD,S]
__global__ __launch_bounds__(256)
void rope_kernel(const h16* __restrict__ qkv, const float* __restrict__ pos,
                 h16* __restrict__ qr, h16* __restrict__ kr, h16* __restrict__ vt) {
    int tid = blockIdx.x * 256 + threadIdx.x;       // B*S*H*32 = 4,194,304
    int p  = tid & 31;
    int h  = (tid >> 5) & 7;
    int s  = (tid >> 8) & 1023;
    int b  = tid >> 18;
    int j  = p & 15, hs = p >> 4;
    int d0 = hs * 32 + 2 * j;
    float freq = __powf(10000.f, -(float)(2 * j) * (1.0f / 32.0f));
    float ang  = pos[((size_t)b * CC * SS + s) * 2 + hs] * freq;  // pos[b,0,s,hs]
    float cs = __cosf(ang), sn = __sinf(ang);
    size_t roff = ((size_t)(b * SS + s)) * (3 * DD);
    float q1 = (float)qkv[roff + h * HD + d0];
    float q2 = (float)qkv[roff + h * HD + d0 + 1];
    float k1 = (float)qkv[roff + DD + h * HD + d0];
    float k2 = (float)qkv[roff + DD + h * HD + d0 + 1];
    size_t ooff = ((size_t)((b * HH + h) * SS + s)) * HD + d0;
    qr[ooff]     = (h16)(q1 * cs - q2 * sn);
    qr[ooff + 1] = (h16)(q1 * sn + q2 * cs);
    kr[ooff]     = (h16)(k1 * cs - k2 * sn);
    kr[ooff + 1] = (h16)(k1 * sn + k2 * cs);
    int vd = 2 * p;
    float v1 = (float)qkv[roff + 2 * DD + h * HD + vd];
    float v2 = (float)qkv[roff + 2 * DD + h * HD + vd + 1];
    vt[((size_t)((b * HH + h) * HD + vd)) * SS + s]     = (h16)v1;
    vt[((size_t)((b * HH + h) * HD + vd + 1)) * SS + s] = (h16)v2;
}

// ---------- attention: one block = (b,h) x 32 query rows; scores panel in LDS ----------
#define SSTR 1032   // f32 score row stride
#define PSTR 1040   // f16 P row stride
#define ATT_SMEM (32*SSTR*4 + 32*PSTR*2)   // 132096 + 66560 = 198656 bytes

__global__ __launch_bounds__(256)
void attn_kernel(const h16* __restrict__ qr, const h16* __restrict__ kr,
                 const h16* __restrict__ vt, h16* __restrict__ out) {
    extern __shared__ char smem[];
    float* sS = (float*)smem;                        // [32][SSTR]
    h16*   sP = (h16*)(smem + 32 * SSTR * 4);        // [32][PSTR]
    const int t = threadIdx.x, lane = t & 31, wid = t >> 5;
    const int bh = blockIdx.y;                       // b*H + h
    const int q0 = blockIdx.x * 32;
    const h16* qb = qr + (size_t)bh * (SS * HD);
    const h16* kb = kr + (size_t)bh * (SS * HD);
    const h16* vb = vt + (size_t)bh * (HD * SS);
    const int akb  = ((lane >> 4) << 3);
    const int bkb  = ((lane >> 4) << 3);
    const int lrow = lane & 15;
    const int hi8  = ((lane >> 4) << 3);

    // ---- phase 1: scores = (q @ k^T) / 8   (2 x 64 tiles of 16x16, K=HD=64)
    for (int tile = wid; tile < 128; tile += 8) {
        int qm = tile & 1, kn = tile >> 1;
        v8f acc = {};
#pragma unroll
        for (int c = 0; c < 2; ++c) {
            v16h af = load_afrag(qb + (size_t)(q0 + qm * 16 + lrow) * HD + c * 32 + akb);
            v16h bf = load_bfrag(kb + (size_t)(kn * 16 + lrow) * HD + c * 32 + bkb);
            acc = wmma_f16(af, bf, acc);
        }
        int col = kn * 16 + lrow;
#pragma unroll
        for (int r = 0; r < 8; ++r)
            sS[(qm * 16 + r + hi8) * SSTR + col] = acc[r] * 0.125f;  // 1/sqrt(64)
    }
    __syncthreads();

    // ---- phase 2: softmax, 4 rows per wave, write P (f16)
    for (int rr = 0; rr < 4; ++rr) {
        int row = wid * 4 + rr;
        float* sr = sS + row * SSTR;
        float mx = -3.0e38f;
#pragma unroll
        for (int i = 0; i < 32; ++i) mx = fmaxf(mx, sr[lane + i * 32]);
#pragma unroll
        for (int o = 16; o; o >>= 1) mx = fmaxf(mx, __shfl_xor(mx, o, 32));
        float sum = 0.f;
#pragma unroll
        for (int i = 0; i < 32; ++i) sum += __expf(sr[lane + i * 32] - mx);
#pragma unroll
        for (int o = 16; o; o >>= 1) sum += __shfl_xor(sum, o, 32);
        float inv = 1.0f / sum;
        h16* pr = sP + row * PSTR;
#pragma unroll
        for (int i = 0; i < 32; ++i) {
            int c = lane + i * 32;
            pr[c] = (h16)(__expf(sr[c] - mx) * inv);
        }
    }
    __syncthreads();

    // ---- phase 3: O = P @ V^T'   (8 output tiles of 16x16, one per wave, K=1024)
    int m = wid & 1, n = wid >> 1;
    v8f acc = {};
    for (int kt = 0; kt < 32; ++kt) {
        v16h af = load_afrag(sP + (m * 16 + lrow) * PSTR + kt * 32 + akb);
        v16h bf = load_bfrag(vb + (size_t)(n * 16 + lrow) * SS + kt * 32 + bkb);
        acc = wmma_f16(af, bf, acc);
    }
    int b = bh >> 3, h = bh & 7;
    int col = h * HD + n * 16 + lrow;
#pragma unroll
    for (int r = 0; r < 8; ++r) {
        int qrow = q0 + m * 16 + r + hi8;
        out[((size_t)(b * SS + qrow)) * DD + col] = (h16)acc[r];
    }
}

// ---------- launch ----------
extern "C" void kernel_launch(void* const* d_in, const int* in_sizes, int n_in,
                              void* d_out, int out_size, void* d_ws, size_t ws_size,
                              hipStream_t stream) {
    const float* x    = (const float*)d_in[0];
    const float* pos  = (const float*)d_in[1];
    const float* Wqkv = (const float*)d_in[2];
    const float* bqkv = (const float*)d_in[3];
    const float* Wo   = (const float*)d_in[4];
    const float* bo   = (const float*)d_in[5];
    const float* ln1g = (const float*)d_in[6];
    const float* ln1b = (const float*)d_in[7];
    const float* ln2g = (const float*)d_in[8];
    const float* ln2b = (const float*)d_in[9];
    const float* W1   = (const float*)d_in[10];
    const float* bf1  = (const float*)d_in[11];
    const float* W2   = (const float*)d_in[12];
    const float* bf2  = (const float*)d_in[13];
    float* out = (float*)d_out;
    char*  ws  = (char*)d_ws;

    // workspace layout (bytes), overlapped regions
    const size_t O_WQKVT = 0;                            // 1536*512*2 = 1,572,864
    const size_t O_WOT   = O_WQKVT + 1572864;            // 512*512*2  =   524,288
    const size_t O_W1T   = O_WOT   + 524288;             // 2048*512*2 = 2,097,152
    const size_t O_W2T   = O_W1T   + 2097152;            // 512*2048*2 = 2,097,152
    const size_t O_A     = O_W2T   + 2097152;            // region A (64MB)
    const size_t O_LN1   = O_A;                          // 16MB (dead after QKV gemm)
    const size_t O_QKV   = O_A + 16777216;               // 48MB (dead after rope)
    const size_t O_FF1   = O_A;                          // 64MB overlays ln1+qkv
    const size_t O_B     = O_A + 67108864;               // region B (48MB)
    const size_t O_QR    = O_B;
    const size_t O_KR    = O_B + 16777216;
    const size_t O_VT    = O_B + 33554432;
    const size_t O_LN2   = O_B;                          // overlays qr after attention
    const size_t O_ATT   = O_B + 50331648;               // 16MB
    const size_t O_SRC   = O_ATT + 16777216;             // 32MB f32

    h16*  WqkvT = (h16*)(ws + O_WQKVT);
    h16*  WoT   = (h16*)(ws + O_WOT);
    h16*  W1T   = (h16*)(ws + O_W1T);
    h16*  W2T   = (h16*)(ws + O_W2T);
    h16*  ln1   = (h16*)(ws + O_LN1);
    h16*  qkv   = (h16*)(ws + O_QKV);
    h16*  ff1   = (h16*)(ws + O_FF1);
    h16*  qrp   = (h16*)(ws + O_QR);
    h16*  krp   = (h16*)(ws + O_KR);
    h16*  vtp   = (h16*)(ws + O_VT);
    h16*  ln2   = (h16*)(ws + O_LN2);
    h16*  att   = (h16*)(ws + O_ATT);
    float* src  = (float*)(ws + O_SRC);

    const long XCH = (long)CC * SS * DD;   // batch stride of x / d_out (channel-0 view)
    const long SCH = (long)SS * DD;        // batch stride of contiguous [M,D] f32

    // 0) pass-through copy of all channels; channel 0 overwritten at the end
    (void)hipMemcpyAsync(d_out, d_in[0], (size_t)BB * CC * SS * DD * 4,
                         hipMemcpyDeviceToDevice, stream);

    // 1) weights -> f16 transposed
    wcvt_kernel<<<(512 * 1536 + 255) / 256, 256, 0, stream>>>(Wqkv, WqkvT, 512, 1536);
    wcvt_kernel<<<(512 * 512  + 255) / 256, 256, 0, stream>>>(Wo,   WoT,   512, 512);
    wcvt_kernel<<<(512 * 2048 + 255) / 256, 256, 0, stream>>>(W1,   W1T,   512, 2048);
    wcvt_kernel<<<(2048 * 512 + 255) / 256, 256, 0, stream>>>(W2,   W2T,   2048, 512);

    // 2) LN1 on x[:,0]
    ln_kernel<<<MTOT / 8, 256, 0, stream>>>(x, XCH, ln1g, ln1b, ln1);

    // 3) QKV = ln1 @ Wqkv + bqkv  -> f16 [M,1536]
    gemm_wmma<EPI_F16><<<dim3(1536 / TN, MTOT / TM), 256, 0, stream>>>(
        ln1, WqkvT, bqkv, nullptr, 0, 0, qkv, nullptr, MTOT, 1536, 512);

    // 4) RoPE + split + V transpose
    rope_kernel<<<(BB * SS * HH * 32) / 256, 256, 0, stream>>>(qkv, pos, qrp, krp, vtp);

    // 5) attention
    attn_kernel<<<dim3(SS / 32, BB * HH), 256, ATT_SMEM, stream>>>(qrp, krp, vtp, att);

    // 6) src = x[:,0] + att @ Wo + bo  -> f32 [M,512]
    gemm_wmma<EPI_RES_F32><<<dim3(512 / TN, MTOT / TM), 256, 0, stream>>>(
        att, WoT, bo, x, XCH, SCH, nullptr, src, MTOT, 512, 512);

    // 7) LN2 on src
    ln_kernel<<<MTOT / 8, 256, 0, stream>>>(src, SCH, ln2g, ln2b, ln2);

    // 8) ff1 = gelu(ln2 @ W1 + bf1) -> f16 [M,2048]
    gemm_wmma<EPI_GELU_F16><<<dim3(2048 / TN, MTOT / TM), 256, 0, stream>>>(
        ln2, W1T, bf1, nullptr, 0, 0, ff1, nullptr, MTOT, 2048, 512);

    // 9) out[:,0] = src + ff1 @ W2 + bf2  (scatter into d_out channel 0)
    gemm_wmma<EPI_RES_F32><<<dim3(512 / TN, MTOT / TM), 256, 0, stream>>>(
        ff1, W2T, bf2, src, SCH, XCH, nullptr, out, MTOT, 512, 2048);
}
